// CrossAttentionAdapter_44951127720526
// MI455X (gfx1250) — compile-verified, matrix operand order checked by reading, not run on hardware
//
#include <hip/hip_runtime.h>
#include <hip/hip_bf16.h>
#include <stdint.h>

// Problem constants (match reference: B, CLIP, P, E, H = 1024, 1024, 32, 2048, 16)
static constexpr int Bn    = 1024;   // batch
static constexpr int CLIPd = 1024;   // image-embedding dim
static constexpr int Pd    = 32;     // prefix slots
static constexpr int Ed    = 2048;   // embed dim

typedef __attribute__((ext_vector_type(16))) __bf16 v16bf;
typedef __attribute__((ext_vector_type(8)))  __bf16 v8bf;
typedef __attribute__((ext_vector_type(8)))  float  v8f;
typedef __attribute__((ext_vector_type(4)))  float  v4f;   // native vector for NT stores
typedef __attribute__((ext_vector_type(2)))  uint32_t v2u;

__device__ __forceinline__ uint32_t f32_to_bf16_bits(float x) {
    union { float f; uint32_t u; } c; c.f = x;
    return (c.u + 0x7fffu + ((c.u >> 16) & 1u)) >> 16;   // RNE
}

// ---------------------------------------------------------------------------
// f32 -> bf16 conversion, 4 elements/thread (16-byte in, 8-byte packed out)
// ---------------------------------------------------------------------------
__global__ void __launch_bounds__(256)
cvt_f32_bf16_x4(const v4f* __restrict__ src, v2u* __restrict__ dst, int n4) {
    int i = blockIdx.x * blockDim.x + threadIdx.x;
    int stride = gridDim.x * blockDim.x;
    for (; i < n4; i += stride) {
        v4f v = src[i];
        v2u o;
        o.x = f32_to_bf16_bits(v.x) | (f32_to_bf16_bits(v.y) << 16);
        o.y = f32_to_bf16_bits(v.z) | (f32_to_bf16_bits(v.w) << 16);
        dst[i] = o;
    }
}

// ---------------------------------------------------------------------------
// Fragment load for a 16xK-slice operand stored row-major, K contiguous.
// CDNA5 16-bit A layout (ISA 7.12.2): lanes 0-15 hold rows 0-15 with
// K = {0..7, 16..23}; lanes 16-31 hold rows 0-15 with K = {8..15, 24..31}.
// B operand (= W^T, K x 16) is symmetric, so W's rows (K-contiguous) load
// with the identical pattern.
// ---------------------------------------------------------------------------
__device__ __forceinline__ v16bf
load_frag(const __bf16* __restrict__ base, int ld, int row0, int k0, int lane) {
    int r  = lane & 15;
    int hi = lane >> 4;                 // 0: K+0 / K+16, 1: K+8 / K+24
    const __bf16* p = base + (size_t)(row0 + r) * (size_t)ld + k0 + hi * 8;
    v8bf lo = *reinterpret_cast<const v8bf*>(p);        // 16-byte load
    v8bf hh = *reinterpret_cast<const v8bf*>(p + 16);   // 16-byte load
    return __builtin_shufflevector(lo, hh, 0, 1, 2, 3, 4, 5, 6, 7,
                                           8, 9, 10, 11, 12, 13, 14, 15);
}

// ---------------------------------------------------------------------------
// C[M,N] = A[M,K] * W[N,K]^T + bias[N]
// A, W: bf16 row-major (K contiguous). Output: f32 (Cf) or bf16 (Cb).
// Block = 256 threads = 8 wave32s arranged 2(M) x 4(N); block tile 64 x 256.
// Each wave computes a 32x64 tile = 2x4 (16x16) WMMA accumulators:
// 8 WMMAs per K-step from 2 A-frags + 4 B-frags (1.5 b128 loads / WMMA).
// ---------------------------------------------------------------------------
__global__ void __launch_bounds__(256)
gemm_abt_bf16(const __bf16* __restrict__ A, const __bf16* __restrict__ W,
              const float* __restrict__ bias,
              float* __restrict__ Cf, uint16_t* __restrict__ Cb,
              int M, int N, int K) {
    const int lane = threadIdx.x & 31;
    const int wave = threadIdx.x >> 5;
    const int wy = wave & 1;       // 2 waves along M
    const int wx = wave >> 1;      // 4 waves along N
    const int rowBase = blockIdx.y * 64  + wy * 32;
    const int colBase = blockIdx.x * 256 + wx * 64;

    v8f acc[2][4];
#pragma unroll
    for (int i = 0; i < 2; ++i)
#pragma unroll
        for (int j = 0; j < 4; ++j) acc[i][j] = (v8f){};

    for (int k0 = 0; k0 < K; k0 += 32) {
        v16bf a0 = load_frag(A, K, rowBase,      k0, lane);
        v16bf a1 = load_frag(A, K, rowBase + 16, k0, lane);
        v16bf b0 = load_frag(W, K, colBase,      k0, lane);
        v16bf b1 = load_frag(W, K, colBase + 16, k0, lane);
        v16bf b2 = load_frag(W, K, colBase + 32, k0, lane);
        v16bf b3 = load_frag(W, K, colBase + 48, k0, lane);

        if (k0 + 32 < K) {  // warm caches for next K-step (global_prefetch_b8)
            int r = lane & 15, hi = lane >> 4;
            __builtin_prefetch(A + (size_t)(rowBase + r) * (size_t)K + (k0 + 32) + hi * 8, 0, 1);
            __builtin_prefetch(W + (size_t)(colBase + r) * (size_t)K + (k0 + 32) + hi * 8, 0, 1);
            __builtin_prefetch(W + (size_t)(colBase + 32 + r) * (size_t)K + (k0 + 32) + hi * 8, 0, 1);
        }

        acc[0][0] = __builtin_amdgcn_wmma_f32_16x16x32_bf16(false, a0, false, b0, (short)0, acc[0][0], false, false);
        acc[0][1] = __builtin_amdgcn_wmma_f32_16x16x32_bf16(false, a0, false, b1, (short)0, acc[0][1], false, false);
        acc[0][2] = __builtin_amdgcn_wmma_f32_16x16x32_bf16(false, a0, false, b2, (short)0, acc[0][2], false, false);
        acc[0][3] = __builtin_amdgcn_wmma_f32_16x16x32_bf16(false, a0, false, b3, (short)0, acc[0][3], false, false);
        acc[1][0] = __builtin_amdgcn_wmma_f32_16x16x32_bf16(false, a1, false, b0, (short)0, acc[1][0], false, false);
        acc[1][1] = __builtin_amdgcn_wmma_f32_16x16x32_bf16(false, a1, false, b1, (short)0, acc[1][1], false, false);
        acc[1][2] = __builtin_amdgcn_wmma_f32_16x16x32_bf16(false, a1, false, b2, (short)0, acc[1][2], false, false);
        acc[1][3] = __builtin_amdgcn_wmma_f32_16x16x32_bf16(false, a1, false, b3, (short)0, acc[1][3], false, false);
    }

    // C/D layout (ISA 7.12.2): lane l -> column (l&15); VGPR r -> row r (+8 for lanes 16-31)
    const int cn   = lane & 15;
    const int rOff = (lane >> 4) * 8;

#pragma unroll
    for (int i = 0; i < 2; ++i) {
#pragma unroll
        for (int j = 0; j < 4; ++j) {
            const int col = colBase + j * 16 + cn;
            const float bb = bias ? bias[col] : 0.0f;
#pragma unroll
            for (int r = 0; r < 8; ++r) {
                const int row = rowBase + i * 16 + rOff + r;
                const float v = acc[i][j][r] + bb;
                const size_t idx = (size_t)row * (size_t)N + col;
                if (Cf) Cf[idx] = v;
                else    Cb[idx] = (uint16_t)f32_to_bf16_bits(v);
            }
        }
    }
}

// ---------------------------------------------------------------------------
// out[b, p, :] = C4[b, :] for p in [0, P)
// 256 MB streaming write -> non-temporal 16-byte stores (TH_NT), since nothing
// on-device re-reads the output; keeps L2 for GEMM weights/activations.
// ---------------------------------------------------------------------------
__global__ void __launch_bounds__(256)
broadcast_rows(const v4f* __restrict__ src, v4f* __restrict__ dst) {
    const int EV = Ed / 4;
    int i = blockIdx.x * blockDim.x + threadIdx.x;
    if (i >= Bn * EV) return;
    const int b  = i / EV;
    const int e4 = i - b * EV;
    const v4f v = src[i];
    v4f* o = dst + (size_t)b * Pd * EV + e4;
#pragma unroll
    for (int p = 0; p < Pd; ++p)
        __builtin_nontemporal_store(v, o + (size_t)p * EV);
}

// ---------------------------------------------------------------------------
// Launch. Inputs (setup_inputs order):
//  0 image_embs (B,CLIP) 1 Wm (E,CLIP) 2 bm (E) 3 prefix_queries (P,E)
//  4 Win (3E,E)          5 bin (3E)    6 Wo_mha (E,E) 7 bo_mha (E)
//  8 Wo (E,E)            9 bo (E)
// prefix_queries / Wq / Wk / bq / bk are mathematically dead: softmax over a
// singleton axis is identically 1, so attn_out[b,p,:] == v[b,:].
// ---------------------------------------------------------------------------
extern "C" void kernel_launch(void* const* d_in, const int* in_sizes, int n_in,
                              void* d_out, int out_size, void* d_ws, size_t ws_size,
                              hipStream_t stream) {
    const float* img    = (const float*)d_in[0];
    const float* Wm     = (const float*)d_in[1];
    const float* bm     = (const float*)d_in[2];
    const float* Win    = (const float*)d_in[4];
    const float* bin    = (const float*)d_in[5];
    const float* Wo_mha = (const float*)d_in[6];
    const float* bo_mha = (const float*)d_in[7];
    const float* Wo     = (const float*)d_in[8];
    const float* bo     = (const float*)d_in[9];
    float* out = (float*)d_out;

    // Workspace layout (bytes, 256-aligned), total ~50 MB
    char* ws = (char*)d_ws;
    size_t off = 0;
    auto alloc = [&](size_t bytes) { char* p = ws + off; off = (off + bytes + 255) & ~(size_t)255; return p; };
    uint16_t* imgB = (uint16_t*)alloc((size_t)Bn * CLIPd * 2);   // img bf16
    uint16_t* WmB  = (uint16_t*)alloc((size_t)Ed * CLIPd * 2);   // Wm bf16
    uint16_t* WvB  = (uint16_t*)alloc((size_t)Ed * Ed * 2);      // Wv bf16
    uint16_t* WomB = (uint16_t*)alloc((size_t)Ed * Ed * 2);      // Wo_mha bf16
    uint16_t* WoB  = (uint16_t*)alloc((size_t)Ed * Ed * 2);      // Wo bf16
    uint16_t* C1   = (uint16_t*)alloc((size_t)Bn * Ed * 2);      // mem bf16
    uint16_t* C2   = (uint16_t*)alloc((size_t)Bn * Ed * 2);      // v   bf16
    uint16_t* C3   = (uint16_t*)alloc((size_t)Bn * Ed * 2);      // mha bf16
    float*    C4   = (float*)   alloc((size_t)Bn * Ed * 4);      // final row f32
    (void)ws_size; (void)in_sizes; (void)n_in; (void)out_size;

    auto nblk4 = [](int n4) { int b = (n4 + 255) / 256; return b > 4096 ? 4096 : b; };
    const float* Wv = Win + (size_t)2 * Ed * Ed;    // rows [2E, 3E) of Win
    const float* bv = bin + 2 * Ed;

    cvt_f32_bf16_x4<<<nblk4(Bn * CLIPd / 4), 256, 0, stream>>>((const v4f*)img,    (v2u*)imgB, Bn * CLIPd / 4);
    cvt_f32_bf16_x4<<<nblk4(Ed * CLIPd / 4), 256, 0, stream>>>((const v4f*)Wm,     (v2u*)WmB,  Ed * CLIPd / 4);
    cvt_f32_bf16_x4<<<nblk4(Ed * Ed / 4),    256, 0, stream>>>((const v4f*)Wv,     (v2u*)WvB,  Ed * Ed / 4);
    cvt_f32_bf16_x4<<<nblk4(Ed * Ed / 4),    256, 0, stream>>>((const v4f*)Wo_mha, (v2u*)WomB, Ed * Ed / 4);
    cvt_f32_bf16_x4<<<nblk4(Ed * Ed / 4),    256, 0, stream>>>((const v4f*)Wo,     (v2u*)WoB,  Ed * Ed / 4);

    const dim3 gemmGrid(Ed / 256, Bn / 64);   // (8, 16)

    // G1: mem = img @ Wm^T + bm            (M=B, N=E, K=CLIP)
    gemm_abt_bf16<<<gemmGrid, 256, 0, stream>>>(
        (const __bf16*)imgB, (const __bf16*)WmB, bm, nullptr, C1, Bn, Ed, CLIPd);
    // G2: v = mem @ Wv^T + bv              (K=E)
    gemm_abt_bf16<<<gemmGrid, 256, 0, stream>>>(
        (const __bf16*)C1, (const __bf16*)WvB, bv, nullptr, C2, Bn, Ed, Ed);
    // G3: t = v @ Wo_mha^T + bo_mha        (K=E)
    gemm_abt_bf16<<<gemmGrid, 256, 0, stream>>>(
        (const __bf16*)C2, (const __bf16*)WomB, bo_mha, nullptr, C3, Bn, Ed, Ed);
    // G4: row = t @ Wo^T + bo              (K=E), f32 output
    gemm_abt_bf16<<<gemmGrid, 256, 0, stream>>>(
        (const __bf16*)C3, (const __bf16*)WoB, bo, C4, nullptr, Bn, Ed, Ed);

    // out[b, p, :] = row[b, :]
    const int total4 = Bn * (Ed / 4);
    broadcast_rows<<<(total4 + 255) / 256, 256, 0, stream>>>(
        (const v4f*)C4, (v4f*)out);
}
